// MSAColumnAttention_ckpt_33681133535499
// MI455X (gfx1250) — compile-verified, matrix operand order checked by reading, not run on hardware
//
#include <hip/hip_runtime.h>
#include <hip/hip_bf16.h>

typedef __attribute__((ext_vector_type(16))) __bf16 v16bf;
typedef __attribute__((ext_vector_type(8)))  float  v8f;

#define SS  256   // sequence (attention) length
#define II  384   // columns
#define CC  256   // channels
#define DD  256   // H*CH
#define HH  8
#define CHW 32    // c_h

#if __has_builtin(__builtin_amdgcn_global_load_async_to_lds_b128)
#define HAVE_ASYNC_LDS 1
typedef int v4ig __attribute__((vector_size(16)));
typedef __attribute__((address_space(1))) v4ig* as1_v4ip;
typedef __attribute__((address_space(3))) v4ig* as3_v4ip;
#endif

__device__ inline v8f v8f_zero() {
    v8f z = {0.f,0.f,0.f,0.f,0.f,0.f,0.f,0.f};
    return z;
}

// Load a 16(rows) x 32(K) bf16 fragment in WMMA A/B register layout from a
// row-major source (row stride ld elements, 16B-aligned rows).
// Per ISA 7.12.2 the per-lane data is two contiguous 16-byte spans:
//   lanes 0-15 : row = lane,     bytes [0,16) (K0-7)  + [32,48) (K16-23)
//   lanes 16-31: row = lane-16,  bytes [16,32)(K8-15) + [48,64) (K24-31)
__device__ inline v16bf load_frag(const __hip_bfloat16* src, int ld) {
    int lane = threadIdx.x & 31;
    int half = lane >> 4;
    int r    = lane & 15;
    const char* row = (const char*)(src + (size_t)r * ld);
    union { uint4 q[2]; v16bf v; } f;
    f.q[0] = *(const uint4*)(row + (half ? 16 : 0));
    f.q[1] = *(const uint4*)(row + (half ? 48 : 32));
    return f.v;
}

__device__ inline v8f wmma_bf16(v16bf a, v16bf b, v8f c) {
    return __builtin_amdgcn_wmma_f32_16x16x32_bf16(
        /*neg_a=*/false, a, /*neg_b=*/false, b,
        /*c_mod=*/(short)0, c, /*reuse_a=*/false, /*reuse_b=*/false);
}

// Stage `bytes` (multiple of 16) from global to LDS. Uses CDNA5 async
// global->LDS DMA when available (ASYNCcnt-tracked, no VGPR round trip),
// else plain 128-bit copies. Caller must __syncthreads() afterwards.
__device__ inline void stage_to_lds(const void* gsrc, void* ldst, int bytes,
                                    int tid, int nthreads) {
#if defined(HAVE_ASYNC_LDS)
    for (int off = tid * 16; off < bytes; off += nthreads * 16) {
        __builtin_amdgcn_global_load_async_to_lds_b128(
            (as1_v4ip)((const char*)gsrc + off),
            (as3_v4ip)((char*)ldst + off),
            0, 0);
    }
#if __has_builtin(__builtin_amdgcn_s_wait_asynccnt)
    __builtin_amdgcn_s_wait_asynccnt(0);
#else
    asm volatile("s_wait_asynccnt 0x0" ::: "memory");
#endif
#else
    const uint4* src = (const uint4*)gsrc;
    uint4*       dst = (uint4*)ldst;
    for (int j = tid; j < bytes / 16; j += nthreads) dst[j] = src[j];
#endif
}

// ---------------------------------------------------------------------------
// Kernel A: convert weights f32 -> bf16. Wbf = [wq;wk;wv;wg] (1024x256), WoBf.
// ---------------------------------------------------------------------------
__global__ void pack_weights(const float* __restrict__ wq, const float* __restrict__ wk,
                             const float* __restrict__ wv, const float* __restrict__ wg,
                             const float* __restrict__ wo,
                             __hip_bfloat16* __restrict__ Wbf,
                             __hip_bfloat16* __restrict__ WoBf) {
    int idx = blockIdx.x * 256 + threadIdx.x;        // 5 * 65536 total
    if (idx < 65536)            Wbf[idx]            = __float2bfloat16(wq[idx]);
    else if (idx < 131072)      Wbf[idx]            = __float2bfloat16(wk[idx - 65536]);
    else if (idx < 196608)      Wbf[idx]            = __float2bfloat16(wv[idx - 131072]);
    else if (idx < 262144)      Wbf[idx]            = __float2bfloat16(wg[idx - 196608]);
    else if (idx < 327680)      WoBf[idx - 262144]  = __float2bfloat16(wo[idx - 262144]);
}

// ---------------------------------------------------------------------------
// Kernel B: LayerNorm + fused QKVG projection (128 rows x 1024 x 256 GEMM/WG)
// grid = I * 2 (two 128-row s-blocks per column i), block = 256 (8 waves).
// LDS: xln[128][256] bf16 (64KB) + weight chunk [128][256] bf16 (64KB).
// ---------------------------------------------------------------------------
__global__ void ln_qkvg_kernel(const float* __restrict__ m,
                               const float* __restrict__ lnw, const float* __restrict__ lnb,
                               const __hip_bfloat16* __restrict__ Wbf,
                               const float* __restrict__ bg,
                               __hip_bfloat16* __restrict__ q_ws,
                               __hip_bfloat16* __restrict__ k_ws,
                               __hip_bfloat16* __restrict__ v_ws,
                               __hip_bfloat16* __restrict__ g_ws) {
    extern __shared__ char smem[];
    __hip_bfloat16* xln    = (__hip_bfloat16*)smem;                     // 128*256
    __hip_bfloat16* wchunk = (__hip_bfloat16*)(smem + 128 * 256 * 2);   // 128*256

    const int wg   = blockIdx.x;
    const int i    = wg >> 1;
    const int sblk = (wg & 1) * 128;
    const int tid  = threadIdx.x;
    const int wave = tid >> 5, lane = tid & 31;

    // ---- LayerNorm: each wave normalizes 16 rows (one row at a time) ----
    for (int rr = 0; rr < 16; ++rr) {
        int rloc = wave * 16 + rr;
        int s    = sblk + rloc;
        const float* xrow = m + (((size_t)s * II) + i) * CC;
        float vals[8];
        float sum = 0.f, sq = 0.f;
#pragma unroll
        for (int j = 0; j < 8; ++j) {
            float t = xrow[lane * 8 + j];
            vals[j] = t; sum += t; sq += t * t;
        }
#pragma unroll
        for (int mk = 16; mk >= 1; mk >>= 1) {
            sum += __shfl_xor(sum, mk, 32);
            sq  += __shfl_xor(sq,  mk, 32);
        }
        float mu   = sum * (1.f / 256.f);
        float var  = sq * (1.f / 256.f) - mu * mu;
        float rstd = rsqrtf(var + 1e-5f);
#pragma unroll
        for (int j = 0; j < 8; ++j) {
            int c   = lane * 8 + j;
            float y = (vals[j] - mu) * rstd * lnw[c] + lnb[c];
            xln[rloc * 256 + c] = __float2bfloat16(y);
        }
    }
    __syncthreads();

    const float qscale = 0.17677669529663687f; // 1/sqrt(32)
    const int half = lane >> 4, col = lane & 15;

    // ---- N chunks of 128 cols; chunk 0-1: q, 2-3: k, 4-5: v, 6-7: g ----
    for (int chunk = 0; chunk < 8; ++chunk) {
        stage_to_lds(Wbf + (size_t)chunk * 128 * 256, wchunk, 128 * 256 * 2,
                     tid, 256);
        __syncthreads();
        if (chunk < 7)   // hint next chunk toward L2/L0
            __builtin_prefetch((const char*)(Wbf + (size_t)(chunk + 1) * 128 * 256)
                               + tid * 256, 0, 2);

        v8f acc[8];
#pragma unroll
        for (int t = 0; t < 8; ++t) acc[t] = v8f_zero();

        for (int ks = 0; ks < 8; ++ks) {
            v16bf a = load_frag(xln + (wave * 16) * 256 + ks * 32, 256);
            v16bf b[8];
#pragma unroll
            for (int t = 0; t < 8; ++t)
                b[t] = load_frag(wchunk + (t * 16) * 256 + ks * 32, 256);
#pragma unroll
            for (int t = 0; t < 8; ++t)
                acc[t] = wmma_bf16(a, b[t], acc[t]);
        }

        int which = chunk >> 1;   // 0=q 1=k 2=v 3=g
#pragma unroll
        for (int t = 0; t < 8; ++t) {
            int n  = chunk * 128 + t * 16 + col;
            int d  = n & 255;
            int h  = d >> 5, ch = d & 31;
#pragma unroll
            for (int v = 0; v < 8; ++v) {
                int s = sblk + wave * 16 + v + half * 8;
                float val = acc[t][v];
                size_t off = ((((size_t)i * HH) + h) * SS + s) * CHW + ch;
                if (which == 0)      q_ws[off] = __float2bfloat16(val * qscale);
                else if (which == 1) k_ws[off] = __float2bfloat16(val);
                else if (which == 2) v_ws[off] = __float2bfloat16(val);
                else {
                    float gv = val + bg[d];
                    g_ws[off] = __float2bfloat16(1.f / (1.f + __expf(-gv)));
                }
            }
        }
        __syncthreads();
    }
}

// ---------------------------------------------------------------------------
// Kernel C: attention per (i,h). QK^T -> softmax -> PV -> gate.
// grid = I*H (3072), block = 256 (8 waves, 16 q-rows per wave per pass).
// LDS: K (16KB) + V^T (16KB) + bias (1KB) + per-wave probs (64KB).
// ---------------------------------------------------------------------------
__global__ void attn_kernel(const __hip_bfloat16* __restrict__ q_ws,
                            const __hip_bfloat16* __restrict__ k_ws,
                            const __hip_bfloat16* __restrict__ v_ws,
                            const __hip_bfloat16* __restrict__ g_ws,
                            const float* __restrict__ msa_mask,
                            __hip_bfloat16* __restrict__ o_ws) {
    extern __shared__ char smem[];
    __hip_bfloat16* kl    = (__hip_bfloat16*)smem;            // [256][32]
    __hip_bfloat16* vt    = kl + 256 * 32;                    // [32][256]
    float*          biasv = (float*)(vt + 32 * 256);          // [256]
    __hip_bfloat16* probs = (__hip_bfloat16*)(biasv + 256);   // [8][16][256]

    const int ih = blockIdx.x;
    const int i = ih / HH, h = ih % HH;
    const int tid = threadIdx.x, wave = tid >> 5, lane = tid & 31;
    const size_t base = (((size_t)i * HH) + h) * SS * CHW;

    // stage K via async DMA to LDS
    stage_to_lds(k_ws + base, kl, 256 * 32 * 2, tid, 256);
    // stage V transposed -> vt[ch][s] (transpose -> scalar path)
    for (int j = tid; j < 8192; j += 256) {
        int s = j >> 5, c = j & 31;
        vt[c * 256 + s] = v_ws[base + j];
    }
    // mask bias per key
    for (int j = tid; j < 256; j += 256)
        biasv[j] = 1e9f * (msa_mask[(size_t)j * II + i] - 1.0f);
    __syncthreads();

    const int half = lane >> 4, col = lane & 15;

    for (int pass = 0; pass < 2; ++pass) {
        int r0 = pass * 128 + wave * 16;

        // logits: 16 x 256 (K-dim = CH = 32 -> single WMMA step per tile)
        v16bf aq = load_frag(q_ws + base + (size_t)r0 * CHW, CHW);
        v8f lg[16];
#pragma unroll
        for (int g8 = 0; g8 < 2; ++g8) {
            v16bf bk[8];
#pragma unroll
            for (int t = 0; t < 8; ++t)
                bk[t] = load_frag(kl + ((g8 * 8 + t) * 16) * 32, 32);
#pragma unroll
            for (int t = 0; t < 8; ++t)
                lg[g8 * 8 + t] = wmma_bf16(aq, bk[t], v8f_zero());
        }

        // bias + per-row max (row = vgpr slot + 8*half; reduce over 16 lanes)
        float rmax[8];
#pragma unroll
        for (int v = 0; v < 8; ++v) rmax[v] = -1e30f;
#pragma unroll
        for (int t = 0; t < 16; ++t) {
            float b = biasv[t * 16 + col];
#pragma unroll
            for (int v = 0; v < 8; ++v) {
                lg[t][v] += b;
                rmax[v] = fmaxf(rmax[v], lg[t][v]);
            }
        }
#pragma unroll
        for (int v = 0; v < 8; ++v)
#pragma unroll
            for (int mk = 8; mk >= 1; mk >>= 1)
                rmax[v] = fmaxf(rmax[v], __shfl_xor(rmax[v], mk, 32));

        float rsum[8];
#pragma unroll
        for (int v = 0; v < 8; ++v) rsum[v] = 0.f;
#pragma unroll
        for (int t = 0; t < 16; ++t)
#pragma unroll
            for (int v = 0; v < 8; ++v) {
                float e = __expf(lg[t][v] - rmax[v]);
                lg[t][v] = e;
                rsum[v] += e;
            }
#pragma unroll
        for (int v = 0; v < 8; ++v)
#pragma unroll
            for (int mk = 8; mk >= 1; mk >>= 1)
                rsum[v] += __shfl_xor(rsum[v], mk, 32);
        float rinv[8];
#pragma unroll
        for (int v = 0; v < 8; ++v) rinv[v] = 1.f / rsum[v];

        // re-lay probs via per-wave LDS buffer (C/D layout -> row-major)
        __hip_bfloat16* pw = probs + wave * 16 * 256;
#pragma unroll
        for (int t = 0; t < 16; ++t)
#pragma unroll
            for (int v = 0; v < 8; ++v) {
                int r = v + half * 8;
                pw[r * 256 + t * 16 + col] = __float2bfloat16(lg[t][v] * rinv[v]);
            }

        // O = P (16x256) * V (256x32)
        v8f oa[2];
        oa[0] = v8f_zero(); oa[1] = v8f_zero();
        for (int ks = 0; ks < 8; ++ks) {
            v16bf ap  = load_frag(pw + ks * 32, 256);
            v16bf bv0 = load_frag(vt + 0 * 256 + ks * 32, 256);
            v16bf bv1 = load_frag(vt + 16 * 256 + ks * 32, 256);
            oa[0] = wmma_bf16(ap, bv0, oa[0]);
            oa[1] = wmma_bf16(ap, bv1, oa[1]);
        }

        // gate and scatter to [i][s][h*CH+ch]
#pragma unroll
        for (int t = 0; t < 2; ++t)
#pragma unroll
            for (int v = 0; v < 8; ++v) {
                int s  = r0 + v + half * 8;
                int ch = t * 16 + col;
                float g = __bfloat162float(g_ws[base + (size_t)s * CHW + ch]);
                float val = oa[t][v] * g;
                o_ws[(((size_t)i * SS) + s) * DD + h * CHW + ch] = __float2bfloat16(val);
            }
    }
}

// ---------------------------------------------------------------------------
// Kernel D: output projection 128x256x256 per WG, + bo, swap back to (s,i,c).
// grid = I*2, block = 256. LDS: wo bf16 (128KB).
// ---------------------------------------------------------------------------
__global__ void outproj_kernel(const __hip_bfloat16* __restrict__ o_ws,
                               const __hip_bfloat16* __restrict__ WoBf,
                               const float* __restrict__ bo,
                               float* __restrict__ out) {
    extern __shared__ char smem[];
    __hip_bfloat16* wl = (__hip_bfloat16*)smem;   // [256][256]
    const int tid = threadIdx.x, wave = tid >> 5, lane = tid & 31;

    stage_to_lds(WoBf, wl, 256 * 256 * 2, tid, 256);
    __syncthreads();

    const int wg   = blockIdx.x;
    const int i    = wg >> 1;
    const int sblk = (wg & 1) * 128;
    const int r0   = sblk + wave * 16;
    const __hip_bfloat16* arow = o_ws + (((size_t)i * SS) + r0) * DD;
    const int half = lane >> 4, col = lane & 15;

    for (int chunk = 0; chunk < 2; ++chunk) {
        v8f acc[8];
#pragma unroll
        for (int t = 0; t < 8; ++t) acc[t] = v8f_zero();
        for (int ks = 0; ks < 8; ++ks) {
            v16bf a = load_frag(arow + ks * 32, 256);
            v16bf b[8];
#pragma unroll
            for (int t = 0; t < 8; ++t)
                b[t] = load_frag(wl + (chunk * 128 + t * 16) * 256 + ks * 32, 256);
#pragma unroll
            for (int t = 0; t < 8; ++t)
                acc[t] = wmma_bf16(a, b[t], acc[t]);
        }
#pragma unroll
        for (int t = 0; t < 8; ++t) {
            int c = chunk * 128 + t * 16 + col;
            float bb = bo[c];
#pragma unroll
            for (int v = 0; v < 8; ++v) {
                int s = r0 + v + half * 8;
                out[(((size_t)s * II) + i) * CC + c] = acc[t][v] + bb;
            }
        }
    }
}

// ---------------------------------------------------------------------------
extern "C" void kernel_launch(void* const* d_in, const int* in_sizes, int n_in,
                              void* d_out, int out_size, void* d_ws, size_t ws_size,
                              hipStream_t stream) {
    (void)in_sizes; (void)n_in; (void)out_size; (void)ws_size;
    const float* m        = (const float*)d_in[0];
    const float* msa_mask = (const float*)d_in[1];
    const float* ln_w     = (const float*)d_in[2];
    const float* ln_b     = (const float*)d_in[3];
    const float* wq       = (const float*)d_in[4];
    const float* wk       = (const float*)d_in[5];
    const float* wv       = (const float*)d_in[6];
    const float* wg       = (const float*)d_in[7];
    const float* bg       = (const float*)d_in[8];
    const float* wo       = (const float*)d_in[9];
    const float* bo       = (const float*)d_in[10];
    float* out = (float*)d_out;

    // workspace layout
    char* ws = (char*)d_ws;
    size_t off = 0;
    __hip_bfloat16* Wbf  = (__hip_bfloat16*)(ws + off); off += (size_t)1024 * 256 * 2;
    __hip_bfloat16* WoBf = (__hip_bfloat16*)(ws + off); off += (size_t)256 * 256 * 2;
    const size_t qkv_elems = (size_t)II * HH * SS * CHW;   // 384*8*256*32
    __hip_bfloat16* q_ws = (__hip_bfloat16*)(ws + off); off += qkv_elems * 2;
    __hip_bfloat16* k_ws = (__hip_bfloat16*)(ws + off); off += qkv_elems * 2;
    __hip_bfloat16* v_ws = (__hip_bfloat16*)(ws + off); off += qkv_elems * 2;
    __hip_bfloat16* g_ws = (__hip_bfloat16*)(ws + off); off += qkv_elems * 2;
    __hip_bfloat16* o_ws = (__hip_bfloat16*)(ws + off); off += (size_t)II * SS * DD * 2;

    const int smemB = 128 * 256 * 2 * 2;                     // 128 KB
    const int smemC = 256 * 32 * 2 + 32 * 256 * 2 + 256 * 4 + 8 * 16 * 256 * 2; // ~97 KB
    const int smemD = 256 * 256 * 2;                         // 128 KB
    (void)hipFuncSetAttribute((const void*)ln_qkvg_kernel,
                              hipFuncAttributeMaxDynamicSharedMemorySize, smemB);
    (void)hipFuncSetAttribute((const void*)attn_kernel,
                              hipFuncAttributeMaxDynamicSharedMemorySize, smemC);
    (void)hipFuncSetAttribute((const void*)outproj_kernel,
                              hipFuncAttributeMaxDynamicSharedMemorySize, smemD);

    pack_weights<<<1280, 256, 0, stream>>>(wq, wk, wv, wg, wo, Wbf, WoBf);
    ln_qkvg_kernel<<<II * 2, 256, smemB, stream>>>(m, ln_w, ln_b, Wbf, bg,
                                                   q_ws, k_ws, v_ws, g_ws);
    attn_kernel<<<II * HH, 256, smemC, stream>>>(q_ws, k_ws, v_ws, g_ws,
                                                 msa_mask, o_ws);
    outproj_kernel<<<II * 2, 256, smemD, stream>>>(o_ws, WoBf, bo, out);
}
